// ModulatedSparseTransformerCrossBlock_60206851555799
// MI455X (gfx1250) — matrix operand order repaired and checked by reading, not compiled, not run on hardware
//
#include <hip/hip_runtime.h>
#include <cstdint>
#include <cstddef>

// ---------------------------------------------------------------------------
// Types
// ---------------------------------------------------------------------------
typedef __bf16 bf16_t;
typedef __attribute__((ext_vector_type(8)))  float  v8f;
typedef __attribute__((ext_vector_type(16))) bf16_t v16bf;
typedef __attribute__((ext_vector_type(8)))  bf16_t v8bf;
typedef __attribute__((ext_vector_type(4)))  int    v4i;

#define C_DIM 1024
#define L_DIM 2048
#define LC_DIM 512
#define B_DIM 4
#define H_DIM 16

// ---------------------------------------------------------------------------
// Async global->LDS copies (CDNA5 ASYNCcnt path), with sync fallback so the
// file always compiles. 16 bytes per lane per op.
// Builtin signature (decoded from clang diagnostic): takes v4i pointers,
// global source in addrspace(1), LDS dest in addrspace(3).
// ---------------------------------------------------------------------------
#if __has_builtin(__builtin_amdgcn_global_load_async_to_lds_b128) && \
    __has_builtin(__builtin_amdgcn_s_wait_asynccnt)
#define HAVE_ASYNC 1
#else
#define HAVE_ASYNC 0
#endif

#if HAVE_ASYNC
#define AS1 __attribute__((address_space(1)))
#define AS3 __attribute__((address_space(3)))
static __device__ inline void async_cp16(const bf16_t* g, bf16_t* l) {
  __builtin_amdgcn_global_load_async_to_lds_b128((AS1 v4i*)g, (AS3 v4i*)l, 0, 0);
}
#define WAIT_ASYNC(n) __builtin_amdgcn_s_wait_asynccnt(n)
#else
static __device__ inline void async_cp16(const bf16_t* g, bf16_t* l) {
  *(v8bf*)l = *(const v8bf*)g;
}
#define WAIT_ASYNC(n) ((void)0)
#endif

// ---------------------------------------------------------------------------
// WMMA fragment helpers (layouts per CDNA5 ISA 7.12.2, wave32)
// A 16x32 bf16 frag: lane l: m = l%16, g = l/16; k = {8g..8g+7, 16+8g..16+8g+7}
// B 32x16 bf16 frag: lane l: n = l%16, g = l/16; k = {8g..8g+7, 16+8g..16+8g+7}
// C/D f32 frag: VGPR e, lane l: row = e + 8*(l/16), col = l%16.
// ---------------------------------------------------------------------------
static __device__ inline v16bf mk16(v8bf lo, v8bf hi) {
  v16bf r;
#pragma unroll
  for (int e = 0; e < 8; ++e) { r[e] = lo[e]; r[e + 8] = hi[e]; }
  return r;
}
static __device__ inline v16bf ld_afrag(const bf16_t* row, int g) {
  return mk16(*(const v8bf*)(row + 8 * g), *(const v8bf*)(row + 16 + 8 * g));
}
static __device__ inline v16bf ld_bfrag(const bf16_t* row, int g) {
  return mk16(*(const v8bf*)(row + 8 * g), *(const v8bf*)(row + 16 + 8 * g));
}
static __device__ inline v8f wmma_bf16(v16bf a, v16bf b, v8f c) {
  return __builtin_amdgcn_wmma_f32_16x16x32_bf16(false, a, false, b, (short)0, c,
                                                 false, false);
}

// ---------------------------------------------------------------------------
// Small helper kernels
// ---------------------------------------------------------------------------
__global__ void cvt_f32_bf16(const float* __restrict__ in, bf16_t* __restrict__ out,
                             int n) {
  int i = blockIdx.x * blockDim.x + threadIdx.x;
  if (i < n) out[i] = (bf16_t)in[i];
}

// W[K][N] f32 -> W^T[N][K] bf16 (LDS-tiled 32x32 transpose, 256 threads)
__global__ __launch_bounds__(256) void cvt_transpose_k(
    const float* __restrict__ in, bf16_t* __restrict__ out, int K, int N) {
  __shared__ float tile[32][33];
  int k0 = blockIdx.y * 32, n0 = blockIdx.x * 32;
  int tx = threadIdx.x & 31, ty = threadIdx.x >> 5;  // 32 x 8
#pragma unroll
  for (int i = ty; i < 32; i += 8)
    tile[i][tx] = in[(size_t)(k0 + i) * N + n0 + tx];
  __syncthreads();
#pragma unroll
  for (int i = ty; i < 32; i += 8)
    out[(size_t)(n0 + i) * K + k0 + tx] = (bf16_t)tile[tx][i];
}

__global__ void silu_k(const float* __restrict__ in, float* __restrict__ out, int n) {
  int i = blockIdx.x * blockDim.x + threadIdx.x;
  if (i < n) {
    float v = in[i];
    out[i] = v / (1.f + __expf(-v));
  }
}

// ada = silu(mod) @ w_ada + b_ada : [4,1024] x [1024,6144] (tiny)
__global__ void ada_gemm_k(const float* __restrict__ sm, const float* __restrict__ w,
                           const float* __restrict__ b, float* __restrict__ o) {
  int i = blockIdx.x * blockDim.x + threadIdx.x;  // 4*6144
  int bb = i / (6 * C_DIM), n = i % (6 * C_DIM);
  const float* srow = sm + bb * C_DIM;
  float acc = b[n];
  for (int k = 0; k < C_DIM; ++k) acc = fmaf(srow[k], w[(size_t)k * 6 * C_DIM + n], acc);
  o[i] = acc;
}

// ---------------------------------------------------------------------------
// Fused LayerNorm + modulation -> bf16 activations.
// out = LN(x) * (add_one + scale[c]) + shift[c]
// ---------------------------------------------------------------------------
__global__ __launch_bounds__(256) void ln_mod_k(
    const float* __restrict__ x, bf16_t* __restrict__ out,
    const float* __restrict__ scale, const float* __restrict__ shift,
    int tokens_per_sample, int row_stride, float add_one) {
  __shared__ float rs[8], rs2[8];
  int tok = blockIdx.x;
  int t = threadIdx.x, wave = t >> 5, lane = t & 31;
  const float* xp = x + (size_t)tok * C_DIM;
  float v[4], s = 0.f, s2 = 0.f;
#pragma unroll
  for (int i = 0; i < 4; ++i) {
    v[i] = xp[i * 256 + t];
    s += v[i];
    s2 += v[i] * v[i];
  }
#pragma unroll
  for (int off = 1; off < 32; off <<= 1) {
    s += __shfl_xor(s, off, 32);
    s2 += __shfl_xor(s2, off, 32);
  }
  if (lane == 0) { rs[wave] = s; rs2[wave] = s2; }
  __syncthreads();
  float S = 0.f, S2 = 0.f;
#pragma unroll
  for (int w = 0; w < 8; ++w) { S += rs[w]; S2 += rs2[w]; }
  float mu = S * (1.f / C_DIM);
  float var = S2 * (1.f / C_DIM) - mu * mu;
  float rinv = rsqrtf(var + 1e-6f);
  int soff = (tok / tokens_per_sample) * row_stride;
  bf16_t* op = out + (size_t)tok * C_DIM;
#pragma unroll
  for (int i = 0; i < 4; ++i) {
    int c = i * 256 + t;
    float nv = (v[i] - mu) * rinv;
    op[c] = (bf16_t)(nv * (add_one + scale[soff + c]) + shift[soff + c]);
  }
}

// ---------------------------------------------------------------------------
// bf16 WMMA GEMM: out = A[M,K] @ W^T[N,K]^T + bias, async double-buffered.
//   mode 0: store bf16 | mode 1: exact GELU, bf16 | mode 2: res + acc*gate, f32
// Block tile 128x128, K-step 32, 8 waves each owning 32x64.
// LDS rows padded to 40 elems (dword stride 20 -> conflict-free frag reads).
// ---------------------------------------------------------------------------
#define LDA 40
__global__ __launch_bounds__(256) void gemm_bf16_k(
    const bf16_t* __restrict__ A, const bf16_t* __restrict__ Wt,
    const float* __restrict__ bias, int M, int N, int K, int mode,
    float* __restrict__ outf, bf16_t* __restrict__ outb,
    const float* __restrict__ res, const float* __restrict__ gate,
    int tokens_per_sample, int gate_row_stride) {
  __shared__ bf16_t As[2][128 * LDA];
  __shared__ bf16_t Bs[2][128 * LDA];  // [n][k] (weights pre-transposed)
  int m0 = blockIdx.y * 128, n0 = blockIdx.x * 128;
  int t = threadIdx.x, wave = t >> 5, lane = t & 31;
  int ln16 = lane & 15, g = lane >> 4;
  int wm = (wave & 3) * 32;
  int wn = (wave >> 2) * 64;
  v8f acc[2][4] = {};

  int r = t >> 1, c = (t & 1) * 16;  // each thread: one 32B (2x16B) run per tile
  const bf16_t* ga = A + (size_t)(m0 + r) * K + c;
  const bf16_t* gb = Wt + (size_t)(n0 + r) * K + c;

  // prologue: stage 0
  async_cp16(ga, &As[0][r * LDA + c]);
  async_cp16(ga + 8, &As[0][r * LDA + c + 8]);
  async_cp16(gb, &Bs[0][r * LDA + c]);
  async_cp16(gb + 8, &Bs[0][r * LDA + c + 8]);

  int nk = K / 32;
  for (int it = 0; it < nk; ++it) {
    int cur = it & 1;
    if (it + 1 < nk) {  // issue next stage, then wait for current (<=4 pending)
      int k0 = (it + 1) * 32;
      async_cp16(ga + k0, &As[cur ^ 1][r * LDA + c]);
      async_cp16(ga + k0 + 8, &As[cur ^ 1][r * LDA + c + 8]);
      async_cp16(gb + k0, &Bs[cur ^ 1][r * LDA + c]);
      async_cp16(gb + k0 + 8, &Bs[cur ^ 1][r * LDA + c + 8]);
      WAIT_ASYNC(4);
    } else {
      WAIT_ASYNC(0);
    }
    __syncthreads();
    v16bf af[2], bfr[4];
#pragma unroll
    for (int i = 0; i < 2; ++i)
      af[i] = ld_afrag(&As[cur][(wm + 16 * i + ln16) * LDA], g);
#pragma unroll
    for (int j = 0; j < 4; ++j)
      bfr[j] = ld_bfrag(&Bs[cur][(wn + 16 * j + ln16) * LDA], g);
#pragma unroll
    for (int i = 0; i < 2; ++i)
#pragma unroll
      for (int j = 0; j < 4; ++j) acc[i][j] = wmma_bf16(af[i], bfr[j], acc[i][j]);
    __syncthreads();
  }

#pragma unroll
  for (int i = 0; i < 2; ++i)
#pragma unroll
    for (int j = 0; j < 4; ++j) {
      int col = n0 + wn + 16 * j + ln16;
      float bv = bias[col];
#pragma unroll
      for (int e = 0; e < 8; ++e) {
        int row = m0 + wm + 16 * i + e + 8 * g;
        float v = acc[i][j][e] + bv;
        if (mode == 1) v = 0.5f * v * (1.f + erff(v * 0.70710678f));
        size_t idx = (size_t)row * N + col;
        if (mode <= 1) {
          outb[idx] = (bf16_t)v;
        } else {
          float gv = gate ? gate[(row / tokens_per_sample) * gate_row_stride + col]
                          : 1.f;
          outf[idx] = res[idx] + v * gv;
        }
      }
    }
}

// ---------------------------------------------------------------------------
// Flash attention (wave32): grid (Lq/64, H, B), 128 threads = 4 waves.
// Wave-private 16-row Q tile; shared 64x64 K (async-copied, row-major) and
// V^T (padded, channel-major). Online softmax via 16-lane shuffles.
// ---------------------------------------------------------------------------
#define LDV 72  // padded row stride for Vt / P scratch (bank-spread, 16B-aligned)
__global__ __launch_bounds__(128) void flash_attn_k(
    const bf16_t* __restrict__ Q, int q_ts, size_t q_bs,
    const bf16_t* __restrict__ Kp, int k_ts, size_t k_bs,
    const bf16_t* __restrict__ Vp, int v_ts, size_t v_bs,
    bf16_t* __restrict__ O, int o_ts, size_t o_bs, int Lk, float scale) {
  __shared__ bf16_t Kt[64 * 64];       // [pos][ch]
  __shared__ bf16_t Vt[64 * LDV];      // [ch][pos]
  __shared__ bf16_t Pb[4][16 * LDV];   // per-wave P scratch [row][pos]
  int b = blockIdx.z, h = blockIdx.y;
  int t = threadIdx.x, wave = t >> 5, lane = t & 31;
  int ln16 = lane & 15, g = lane >> 4;
  int q0 = blockIdx.x * 64 + wave * 16;

  const bf16_t* qbase = Q + b * q_bs + (size_t)h * 64;
  const bf16_t* kbase = Kp + b * k_bs + (size_t)h * 64;
  const bf16_t* vbase = Vp + b * v_bs + (size_t)h * 64;

  const bf16_t* qr = qbase + (size_t)(q0 + ln16) * q_ts;
  v16bf qf0 = ld_afrag(qr, g);
  v16bf qf1 = ld_afrag(qr + 32, g);

  v8f o[4] = {};
  float mrow[8], lrow[8];
#pragma unroll
  for (int e = 0; e < 8; ++e) { mrow[e] = -1e30f; lrow[e] = 0.f; }

  for (int kt = 0; kt < Lk; kt += 64) {
    // K tile: 512 x 16B chunks, 4 per thread, async straight into LDS
#pragma unroll
    for (int i = 0; i < 4; ++i) {
      int chunk = t * 4 + i, rr = chunk >> 3, cc = (chunk & 7) * 8;
      async_cp16(kbase + (size_t)(kt + rr) * k_ts + cc, Kt + rr * 64 + cc);
    }
    {  // V tile transposed (manual: vector global loads, scatter LDS stores)
      int rr = t >> 1, c0 = (t & 1) * 32;
      const bf16_t* src = vbase + (size_t)(kt + rr) * v_ts + c0;
      v8bf w0 = *(const v8bf*)src;
      v8bf w1 = *(const v8bf*)(src + 8);
      v8bf w2 = *(const v8bf*)(src + 16);
      v8bf w3 = *(const v8bf*)(src + 24);
#pragma unroll
      for (int i = 0; i < 8; ++i) {
        Vt[(c0 + i) * LDV + rr] = w0[i];
        Vt[(c0 + 8 + i) * LDV + rr] = w1[i];
        Vt[(c0 + 16 + i) * LDV + rr] = w2[i];
        Vt[(c0 + 24 + i) * LDV + rr] = w3[i];
      }
    }
    WAIT_ASYNC(0);
    __syncthreads();

    // S = Q K^T (scaled), 4 tiles of 16x16
    v8f s[4];
#pragma unroll
    for (int j = 0; j < 4; ++j) {
      const bf16_t* kr = Kt + (16 * j + ln16) * 64;
      v8f a = {};
      a = wmma_bf16(qf0, ld_bfrag(kr, g), a);
      a = wmma_bf16(qf1, ld_bfrag(kr + 32, g), a);
#pragma unroll
      for (int e = 0; e < 8; ++e) a[e] *= scale;
      s[j] = a;
    }

    // online softmax
    float mnew[8], alpha[8], rsum[8];
#pragma unroll
    for (int e = 0; e < 8; ++e) {
      float mx = fmaxf(fmaxf(s[0][e], s[1][e]), fmaxf(s[2][e], s[3][e]));
#pragma unroll
      for (int off = 1; off < 16; off <<= 1) mx = fmaxf(mx, __shfl_xor(mx, off, 32));
      mnew[e] = fmaxf(mrow[e], mx);
      alpha[e] = __expf(mrow[e] - mnew[e]);
      mrow[e] = mnew[e];
      rsum[e] = 0.f;
    }
#pragma unroll
    for (int j = 0; j < 4; ++j)
#pragma unroll
      for (int e = 0; e < 8; ++e) {
        float pv = __expf(s[j][e] - mnew[e]);
        s[j][e] = pv;
        rsum[e] += pv;
      }
#pragma unroll
    for (int e = 0; e < 8; ++e) {
#pragma unroll
      for (int off = 1; off < 16; off <<= 1) rsum[e] += __shfl_xor(rsum[e], off, 32);
      lrow[e] = lrow[e] * alpha[e] + rsum[e];
    }
#pragma unroll
    for (int j = 0; j < 4; ++j)
#pragma unroll
      for (int e = 0; e < 8; ++e) o[j][e] *= alpha[e];

    // P (C-layout) -> per-wave LDS -> A-frags (same-wave LDS is in-order)
    bf16_t* pb = &Pb[wave][0];
#pragma unroll
    for (int j = 0; j < 4; ++j)
#pragma unroll
      for (int e = 0; e < 8; ++e)
        pb[(e + 8 * g) * LDV + 16 * j + ln16] = (bf16_t)s[j][e];
    const bf16_t* pr = pb + ln16 * LDV;
    v16bf pf0 = ld_afrag(pr, g);
    v16bf pf1 = ld_afrag(pr + 32, g);

    // O += P V
#pragma unroll
    for (int j = 0; j < 4; ++j) {
      const bf16_t* vr = Vt + (16 * j + ln16) * LDV;
      o[j] = wmma_bf16(pf0, ld_bfrag(vr, g), o[j]);
      o[j] = wmma_bf16(pf1, ld_bfrag(vr + 32, g), o[j]);
    }
    __syncthreads();
  }

  float inv[8];
#pragma unroll
  for (int e = 0; e < 8; ++e) inv[e] = 1.f / lrow[e];
  bf16_t* obase = O + b * o_bs + (size_t)h * 64;
#pragma unroll
  for (int j = 0; j < 4; ++j)
#pragma unroll
    for (int e = 0; e < 8; ++e) {
      int row = q0 + e + 8 * g, col = 16 * j + ln16;
      obase[(size_t)row * o_ts + col] = (bf16_t)(o[j][e] * inv[e]);
    }
}

// ---------------------------------------------------------------------------
// Host-side orchestration
// ---------------------------------------------------------------------------
extern "C" void kernel_launch(void* const* d_in, const int* in_sizes, int n_in,
                              void* d_out, int out_size, void* d_ws, size_t ws_size,
                              hipStream_t stream) {
  const float* x       = (const float*)d_in[0];
  const float* mod     = (const float*)d_in[1];
  const float* context = (const float*)d_in[2];
  const float* w_ada   = (const float*)d_in[3];
  const float* b_ada   = (const float*)d_in[4];
  const float* w_qkv   = (const float*)d_in[5];
  const float* b_qkv   = (const float*)d_in[6];
  const float* w_so    = (const float*)d_in[7];
  const float* b_so    = (const float*)d_in[8];
  const float* g_n2    = (const float*)d_in[9];
  const float* b_n2    = (const float*)d_in[10];
  const float* w_cq    = (const float*)d_in[11];
  const float* b_cq    = (const float*)d_in[12];
  const float* w_ckv   = (const float*)d_in[13];
  const float* b_ckv   = (const float*)d_in[14];
  const float* w_co    = (const float*)d_in[15];
  const float* b_co    = (const float*)d_in[16];
  const float* w_m1    = (const float*)d_in[17];
  const float* b_m1    = (const float*)d_in[18];
  const float* w_m2    = (const float*)d_in[19];
  const float* b_m2    = (const float*)d_in[20];
  float* out = (float*)d_out;

  const int M = B_DIM * L_DIM;  // 8192 tokens
  char* p = (char*)d_ws;
  auto carve = [&](size_t bytes) -> void* {
    void* r = (void*)p;
    p += (bytes + 255) & ~size_t(255);
    return r;
  };
  float*  silu_m = (float*)carve((size_t)B_DIM * C_DIM * 4);
  float*  ada    = (float*)carve((size_t)B_DIM * 6 * C_DIM * 4);
  float*  x1     = (float*)carve((size_t)M * C_DIM * 4);
  bf16_t* hbuf   = (bf16_t*)carve((size_t)M * C_DIM * 2);
  bf16_t* qkvb   = (bf16_t*)carve((size_t)M * 3 * C_DIM * 2);
  bf16_t* qbuf   = (bf16_t*)carve((size_t)M * C_DIM * 2);
  bf16_t* kvb    = (bf16_t*)carve((size_t)B_DIM * LC_DIM * 2 * C_DIM * 2);
  bf16_t* ctxb   = (bf16_t*)carve((size_t)B_DIM * LC_DIM * C_DIM * 2);
  bf16_t* attnb  = (bf16_t*)carve((size_t)M * C_DIM * 2);
  bf16_t* mlph   = (bf16_t*)carve((size_t)M * 4 * C_DIM * 2);
  bf16_t* wqkvb  = (bf16_t*)carve((size_t)C_DIM * 3 * C_DIM * 2);
  bf16_t* wsob   = (bf16_t*)carve((size_t)C_DIM * C_DIM * 2);
  bf16_t* wcqb   = (bf16_t*)carve((size_t)C_DIM * C_DIM * 2);
  bf16_t* wckvb  = (bf16_t*)carve((size_t)C_DIM * 2 * C_DIM * 2);
  bf16_t* wcob   = (bf16_t*)carve((size_t)C_DIM * C_DIM * 2);
  bf16_t* wm1b   = (bf16_t*)carve((size_t)C_DIM * 4 * C_DIM * 2);
  bf16_t* wm2b   = (bf16_t*)carve((size_t)4 * C_DIM * C_DIM * 2);

  // Weight conversion + transpose to [N][K] bf16 (one-time per call, tiny cost)
  auto cvtT = [&](const float* src, bf16_t* dst, int K, int N) {
    cvt_transpose_k<<<dim3(N / 32, K / 32), 256, 0, stream>>>(src, dst, K, N);
  };
  cvtT(w_qkv, wqkvb, C_DIM, 3 * C_DIM);
  cvtT(w_so, wsob, C_DIM, C_DIM);
  cvtT(w_cq, wcqb, C_DIM, C_DIM);
  cvtT(w_ckv, wckvb, C_DIM, 2 * C_DIM);
  cvtT(w_co, wcob, C_DIM, C_DIM);
  cvtT(w_m1, wm1b, C_DIM, 4 * C_DIM);
  cvtT(w_m2, wm2b, 4 * C_DIM, C_DIM);
  cvt_f32_bf16<<<(B_DIM * LC_DIM * C_DIM + 255) / 256, 256, 0, stream>>>(
      context, ctxb, B_DIM * LC_DIM * C_DIM);

  // adaLN: silu(mod) @ w_ada + b_ada
  silu_k<<<(B_DIM * C_DIM + 255) / 256, 256, 0, stream>>>(mod, silu_m, B_DIM * C_DIM);
  ada_gemm_k<<<(B_DIM * 6 * C_DIM) / 256, 256, 0, stream>>>(silu_m, w_ada, b_ada, ada);

  // ---- MSA ----
  ln_mod_k<<<M, 256, 0, stream>>>(x, hbuf, ada + C_DIM, ada, L_DIM, 6 * C_DIM, 1.f);
  gemm_bf16_k<<<dim3(3 * C_DIM / 128, M / 128), 256, 0, stream>>>(
      hbuf, wqkvb, b_qkv, M, 3 * C_DIM, C_DIM, 0, nullptr, qkvb, nullptr, nullptr,
      L_DIM, 0);
  flash_attn_k<<<dim3(L_DIM / 64, H_DIM, B_DIM), 128, 0, stream>>>(
      qkvb, 3 * C_DIM, (size_t)L_DIM * 3 * C_DIM,
      qkvb + C_DIM, 3 * C_DIM, (size_t)L_DIM * 3 * C_DIM,
      qkvb + 2 * C_DIM, 3 * C_DIM, (size_t)L_DIM * 3 * C_DIM,
      attnb, C_DIM, (size_t)L_DIM * C_DIM, L_DIM, 0.125f);
  gemm_bf16_k<<<dim3(C_DIM / 128, M / 128), 256, 0, stream>>>(
      attnb, wsob, b_so, M, C_DIM, C_DIM, 2, x1, nullptr, x, ada + 2 * C_DIM, L_DIM,
      6 * C_DIM);

  // ---- MCA ----
  ln_mod_k<<<M, 256, 0, stream>>>(x1, hbuf, g_n2, b_n2, L_DIM, 0, 0.f);
  gemm_bf16_k<<<dim3(C_DIM / 128, M / 128), 256, 0, stream>>>(
      hbuf, wcqb, b_cq, M, C_DIM, C_DIM, 0, nullptr, qbuf, nullptr, nullptr, L_DIM, 0);
  gemm_bf16_k<<<dim3(2 * C_DIM / 128, (B_DIM * LC_DIM) / 128), 256, 0, stream>>>(
      ctxb, wckvb, b_ckv, B_DIM * LC_DIM, 2 * C_DIM, C_DIM, 0, nullptr, kvb, nullptr,
      nullptr, L_DIM, 0);
  flash_attn_k<<<dim3(L_DIM / 64, H_DIM, B_DIM), 128, 0, stream>>>(
      qbuf, C_DIM, (size_t)L_DIM * C_DIM,
      kvb, 2 * C_DIM, (size_t)LC_DIM * 2 * C_DIM,
      kvb + C_DIM, 2 * C_DIM, (size_t)LC_DIM * 2 * C_DIM,
      attnb, C_DIM, (size_t)L_DIM * C_DIM, LC_DIM, 0.125f);
  gemm_bf16_k<<<dim3(C_DIM / 128, M / 128), 256, 0, stream>>>(
      attnb, wcob, b_co, M, C_DIM, C_DIM, 2, x1, nullptr, x1, nullptr, L_DIM, 0);

  // ---- FFN ----
  ln_mod_k<<<M, 256, 0, stream>>>(x1, hbuf, ada + 4 * C_DIM, ada + 3 * C_DIM, L_DIM,
                                  6 * C_DIM, 1.f);
  gemm_bf16_k<<<dim3(4 * C_DIM / 128, M / 128), 256, 0, stream>>>(
      hbuf, wm1b, b_m1, M, 4 * C_DIM, C_DIM, 1, nullptr, mlph, nullptr, nullptr,
      L_DIM, 0);
  gemm_bf16_k<<<dim3(C_DIM / 128, M / 128), 256, 0, stream>>>(
      mlph, wm2b, b_m2, M, C_DIM, 4 * C_DIM, 2, out, nullptr, x1, ada + 5 * C_DIM,
      L_DIM, 6 * C_DIM);
}
